// HybridRNN_74689481277981
// MI455X (gfx1250) — matrix operand order repaired and checked by reading, not compiled, not run on hardware
//
#include <hip/hip_runtime.h>
#include <hip/hip_bf16.h>

typedef __attribute__((ext_vector_type(16))) __bf16 v16bf;
typedef __attribute__((ext_vector_type(8)))  __bf16 v8bf;
typedef __attribute__((ext_vector_type(8)))  float  v8f;

#define BB   8192
#define ZZ   79
#define FF   32
#define CC   256
#define MT   2              // M tiles (of 16 rows) per workgroup
#define LROW 264            // padded LDS row stride (bf16 elems): 528B -> conflict-free b128 loads

static __device__ __forceinline__ v16bf lds_afrag(const __bf16* hrow) {
    const v8bf alo = *(const v8bf*)hrow;
    const v8bf ahi = *(const v8bf*)(hrow + 16);
    return __builtin_shufflevector(alo, ahi, 0,1,2,3,4,5,6,7,8,9,10,11,12,13,14,15);
}

__global__ __launch_bounds__(512, 1)
void hybrid_rnn_wmma(const float* __restrict__ x,  const float* __restrict__ Wx,
                     const float* __restrict__ Wh, const float* __restrict__ b,
                     const float* __restrict__ Wd, const float* __restrict__ bd,
                     float* __restrict__ out)
{
    const int tid  = threadIdx.x;
    const int wave = tid >> 5;        // 0..15 : N-tile (16 output columns)
    const int lane = tid & 31;
    const int ln   = lane & 15;
    const int hi   = lane >> 4;       // half-wave select per WMMA layouts
    const int col  = wave * 16 + ln;  // this lane's output column (0..255)
    const int rowWG = blockIdx.x * (MT * 16);

    __shared__ __align__(16) __bf16 hbuf[2][MT * 16][LROW];

    // ---- Preload per-wave weight column slices into registers (B-fragments).
    // B 32x16 bf16 layout: lane = N, element e -> K = 8*hi + e + (e & 8).
    v16bf bwx;
#pragma unroll
    for (int e = 0; e < 16; ++e)
        bwx[e] = (__bf16)Wx[(8 * hi + e + (e & 8)) * CC + col];
    // Pin in VGPRs: opaque asm prevents per-iteration rematerialization of the
    // global loads + converts.
    asm volatile("" : "+v"(bwx));

    v16bf bwh[8];
#pragma unroll
    for (int c = 0; c < 8; ++c) {
#pragma unroll
        for (int e = 0; e < 16; ++e)
            bwh[c][e] = (__bf16)Wh[(32 * c + 8 * hi + e + (e & 8)) * CC + col];
        asm volatile("" : "+v"(bwh[c]));
    }

    const float bias_b = b[col];

    // Per-m x pointers for the reversed scan; decremented by FF each step
    // (strength-reduced addressing: one 64-bit add per step instead of mul).
    const float* xptr[MT];
#pragma unroll
    for (int m = 0; m < MT; ++m)
        xptr[m] = x + (size_t)(rowWG + m * 16 + ln) * (ZZ * FF) + (size_t)(ZZ - 1) * FF;

    // ---- Recurrence: t = 0..Z-1 over reversed sequence (go_backwards)
    int cur = 0;
    for (int t = 0; t < ZZ; ++t) {
        const int nxt = cur ^ 1;
#pragma unroll
        for (int m = 0; m < MT; ++m) {
            // A-fragment of x slice: 16x32 bf16. Lane holds row M=ln;
            // elems 0..7 -> K = 8*hi..+7 ; elems 8..15 -> K = 16+8*hi..+7
            const float* xrow = xptr[m];
            const float4 p0 = *(const float4*)(xrow + 8 * hi);
            const float4 p1 = *(const float4*)(xrow + 8 * hi + 4);
            const float4 q0 = *(const float4*)(xrow + 16 + 8 * hi);
            const float4 q1 = *(const float4*)(xrow + 16 + 8 * hi + 4);
            // Prefetch next step's x slice one barrier ahead (global_prefetch_b8).
            if (t < ZZ - 1) __builtin_prefetch(xrow - FF, 0, 1);

            v16bf ax;
            ax[0]=(__bf16)p0.x; ax[1]=(__bf16)p0.y; ax[2]=(__bf16)p0.z; ax[3]=(__bf16)p0.w;
            ax[4]=(__bf16)p1.x; ax[5]=(__bf16)p1.y; ax[6]=(__bf16)p1.z; ax[7]=(__bf16)p1.w;
            ax[8]=(__bf16)q0.x; ax[9]=(__bf16)q0.y; ax[10]=(__bf16)q0.z; ax[11]=(__bf16)q0.w;
            ax[12]=(__bf16)q1.x; ax[13]=(__bf16)q1.y; ax[14]=(__bf16)q1.z; ax[15]=(__bf16)q1.w;

            // Two independent accumulator chains -> 2 WMMAs in flight per tile
            v8f acc0;
#pragma unroll
            for (int j = 0; j < 8; ++j) acc0[j] = bias_b;
            acc0 = __builtin_amdgcn_wmma_f32_16x16x32_bf16(
                       false, ax, false, bwx, (short)0, acc0, false, false);

            if (t > 0) {
                const __bf16* hrowb = &hbuf[cur][m * 16 + ln][8 * hi];
                v8f acc1 = {};
#pragma unroll
                for (int c = 0; c < 4; ++c) {
                    const v16bf a0 = lds_afrag(hrowb + 32 * (2 * c));
                    const v16bf a1 = lds_afrag(hrowb + 32 * (2 * c + 1));
                    acc0 = __builtin_amdgcn_wmma_f32_16x16x32_bf16(
                               false, a0, false, bwh[2 * c],     (short)0, acc0, false, false);
                    acc1 = __builtin_amdgcn_wmma_f32_16x16x32_bf16(
                               false, a1, false, bwh[2 * c + 1], (short)0, acc1, false, false);
                }
#pragma unroll
                for (int j = 0; j < 8; ++j) acc0[j] += acc1[j];
            }

            // relu + write new h tile (C/D layout: acc[j] -> row j+8*hi, col)
#pragma unroll
            for (int j = 0; j < 8; ++j) {
                float v = __builtin_fmaxf(acc0[j], 0.0f);
                hbuf[nxt][m * 16 + 8 * hi + j][col] = (__bf16)v;
            }
            xptr[m] -= FF;
        }
        __syncthreads();
        cur ^= 1;
    }

    // ---- Dense head: out = relu(h_last @ Wd + bd)  (one-time; remat harmless)
    v16bf bwd[8];
#pragma unroll
    for (int c = 0; c < 8; ++c)
#pragma unroll
        for (int e = 0; e < 16; ++e)
            bwd[c][e] = (__bf16)Wd[(32 * c + 8 * hi + e + (e & 8)) * CC + col];
    const float bias_d = bd[col];

#pragma unroll
    for (int m = 0; m < MT; ++m) {
        v8f acc0;
#pragma unroll
        for (int j = 0; j < 8; ++j) acc0[j] = bias_d;
        v8f acc1 = {};
        const __bf16* hrowb = &hbuf[cur][m * 16 + ln][8 * hi];
#pragma unroll
        for (int c = 0; c < 4; ++c) {
            const v16bf a0 = lds_afrag(hrowb + 32 * (2 * c));
            const v16bf a1 = lds_afrag(hrowb + 32 * (2 * c + 1));
            acc0 = __builtin_amdgcn_wmma_f32_16x16x32_bf16(
                       false, a0, false, bwd[2 * c],     (short)0, acc0, false, false);
            acc1 = __builtin_amdgcn_wmma_f32_16x16x32_bf16(
                       false, a1, false, bwd[2 * c + 1], (short)0, acc1, false, false);
        }
#pragma unroll
        for (int j = 0; j < 8; ++j) {
            float v = __builtin_fmaxf(acc0[j] + acc1[j], 0.0f);
            out[(size_t)(rowWG + m * 16 + 8 * hi + j) * CC + col] = v;
        }
    }
}

extern "C" void kernel_launch(void* const* d_in, const int* in_sizes, int n_in,
                              void* d_out, int out_size, void* d_ws, size_t ws_size,
                              hipStream_t stream)
{
    const float* x  = (const float*)d_in[0];
    const float* Wx = (const float*)d_in[1];
    const float* Wh = (const float*)d_in[2];
    const float* b  = (const float*)d_in[3];
    const float* Wd = (const float*)d_in[4];
    const float* bd = (const float*)d_in[5];
    float* out = (float*)d_out;

    dim3 grid(BB / (MT * 16));   // 256 workgroups
    dim3 block(512);             // 16 waves (wave32)
    hipLaunchKernelGGL(hybrid_rnn_wmma, grid, block, 0, stream,
                       x, Wx, Wh, b, Wd, bd, out);
}
